// DSRBlock_78529182040557
// MI455X (gfx1250) — matrix operand order repaired and checked by reading, not compiled, not run on hardware
//
#include <hip/hip_runtime.h>
#include <math.h>

// ---------------- constants ----------------
constexpr int   kN  = 4096;   // nodes
constexpr int   kC  = 128;    // in features
constexpr int   kHD = 256;    // HEADS*H
constexpr int   kK  = 64;     // clusters
constexpr float kETA = 0.5f;
constexpr float kRADIUS = 2.0f;

typedef __attribute__((ext_vector_type(16))) __bf16 v16bf;
typedef __attribute__((ext_vector_type(8)))  __bf16 v8bf;
typedef __attribute__((ext_vector_type(8)))  float  v8f;

// ---------------- helpers ----------------
__device__ __forceinline__ float sigm(float x){ return 1.f/(1.f+__expf(-x)); }
__device__ __forceinline__ float lrelu(float x){ return x>0.f?x:0.2f*x; }
__device__ __forceinline__ float eluf(float x){ return x>0.f?x:(__expf(x)-1.f); }
__device__ __forceinline__ float hcgate(float l, float tc){
  float z = sigm(l/tc)*1.2f - 0.1f;               // (zeta-gamma)=1.2, gamma=-0.1
  return fminf(1.f, fmaxf(0.f, z));
}
__device__ __forceinline__ float blockReduceSum(float v, float* sh){
  int tid = threadIdx.x;
  sh[tid]=v; __syncthreads();
  for(int off=128; off>0; off>>=1){ if(tid<off) sh[tid]+=sh[tid+off]; __syncthreads(); }
  float r = sh[0]; __syncthreads();
  return r;
}

// ---------------- generic WMMA GEMM ----------------
// C[M x Ncols] = rowscale[i] * sum_k A[i,k] * kscale[k] * B[k,n]  (+ identity term) (+beta*C)
// grid = (Ncols/64, M/64), block = 256 (8 waves). Kd % 32 == 0.
// LDS pitch 40 halfs (80B): 16B-aligned fragments, conflict-free row access.
template<typename TA>
__global__ __launch_bounds__(256) void k_gemm(
    const TA* __restrict__ A, int lda,
    const float* __restrict__ B, int ldb,
    float* __restrict__ C, int ldc,
    int Kd,
    const float* __restrict__ kscale,
    const float* __restrict__ rowscale,
    int add_identity, int beta)
{
  __shared__ __align__(16) __bf16 As[64][40];   // A tile 64x32, row-major
  __shared__ __align__(16) __bf16 BsT[64][40];  // B tile 32x64 stored transposed: BsT[n][k]
  int tid  = threadIdx.x;
  int n0   = blockIdx.x*64, m0 = blockIdx.y*64;
  int wave = tid>>5, lane = tid&31;
  int wm   = (wave>>1)*16, wn = (wave&1)*32;
  int lh   = lane>>4, ll = lane&15;
  v8f acc0 = {0.f,0.f,0.f,0.f,0.f,0.f,0.f,0.f};
  v8f acc1 = acc0;
  int ar = tid>>2, ac = (tid&3)*8;   // A tile loader: 64 rows x 32 cols, 8/thread
  int br = tid>>3, bc = (tid&7)*8;   // B tile loader: 32 rows x 64 cols, 8/thread
  for (int k0=0; k0<Kd; k0+=32){
    {
      v8bf av;
      #pragma unroll
      for (int u=0; u<8; u++){
        float f = (float)A[(size_t)(m0+ar)*lda + (k0+ac+u)];
        av[u] = (__bf16)f;
      }
      *(v8bf*)&As[ar][ac] = av;      // single ds_store_b128
    }
    {
      float ks = kscale ? kscale[k0+br] : 1.f;
      #pragma unroll
      for (int u=0; u<8; u++){
        float bv = B[(size_t)(k0+br)*ldb + (n0+bc+u)] * ks;
        BsT[bc+u][br] = (__bf16)bv;  // transposed scatter
      }
    }
    __syncthreads();
    // A fragment: 32 contiguous bytes per lane (k = 16*half + s, matches B)
    int arow = wm + ll;
    int kh = lh*16;
    v8bf alo = *(const v8bf*)&As[arow][kh];
    v8bf ahi = *(const v8bf*)&As[arow][kh+8];
    v16bf a = __builtin_shufflevector(alo, ahi, 0,1,2,3,4,5,6,7,8,9,10,11,12,13,14,15);
    // B fragments: 32 contiguous bytes per lane from transposed tile
    int kb = lh*16;
    int c0 = wn + ll;
    v8bf b0l = *(const v8bf*)&BsT[c0][kb];
    v8bf b0h = *(const v8bf*)&BsT[c0][kb+8];
    v16bf b0 = __builtin_shufflevector(b0l, b0h, 0,1,2,3,4,5,6,7,8,9,10,11,12,13,14,15);
    v8bf b1l = *(const v8bf*)&BsT[c0+16][kb];
    v8bf b1h = *(const v8bf*)&BsT[c0+16][kb+8];
    v16bf b1 = __builtin_shufflevector(b1l, b1h, 0,1,2,3,4,5,6,7,8,9,10,11,12,13,14,15);
    acc0 = __builtin_amdgcn_wmma_f32_16x16x32_bf16(false,a,false,b0,(short)0,acc0,false,false);
    acc1 = __builtin_amdgcn_wmma_f32_16x16x32_bf16(false,a,false,b1,(short)0,acc1,false,false);
    __syncthreads();
  }
  #pragma unroll
  for (int r=0; r<8; r++){
    int row = wm + lh*8 + r;
    int gr  = m0 + row;
    float rs = rowscale ? rowscale[gr] : 1.f;
    {
      int gc = n0 + wn + ll;
      float v = acc0[r]*rs;
      if (add_identity){
        float ks2 = kscale ? kscale[gr] : 1.f;
        v += rs*ks2*B[(size_t)gr*ldb+gc];
      }
      size_t ix=(size_t)gr*ldc+gc;
      if (beta) v += C[ix];
      C[ix]=v;
    }
    {
      int gc = n0 + wn + 16 + ll;
      float v = acc1[r]*rs;
      if (add_identity){
        float ks2 = kscale ? kscale[gr] : 1.f;
        v += rs*ks2*B[(size_t)gr*ldb+gc];
      }
      size_t ix=(size_t)gr*ldc+gc;
      if (beta) v += C[ix];
      C[ix]=v;
    }
  }
}

// ---------------- attention prep ----------------
__global__ __launch_bounds__(256) void k_attvec(const float* __restrict__ xh,
    const float* __restrict__ att_src, const float* __restrict__ att_dst,
    float* __restrict__ asrc, float* __restrict__ adst)
{
  int i = blockIdx.x, tid = threadIdx.x;
  __shared__ float sh[256];
  float xv = xh[(size_t)i*kHD + tid];
  sh[tid] = xv*att_src[tid]; __syncthreads();
  for(int off=64; off>0; off>>=1){ if((tid&127)<off) sh[tid]+=sh[tid+off]; __syncthreads(); }
  if(tid==0)   asrc[2*i]   = sh[0];
  if(tid==128) asrc[2*i+1] = sh[128];
  __syncthreads();
  sh[tid] = xv*att_dst[tid]; __syncthreads();
  for(int off=64; off>0; off>>=1){ if((tid&127)<off) sh[tid]+=sh[tid+off]; __syncthreads(); }
  if(tid==0)   adst[2*i]   = sh[0];
  if(tid==128) adst[2*i+1] = sh[128];
}

__device__ __forceinline__ void onmerge(float&m,float&s,float m2,float s2){
  if (s2==0.f) return;
  if (s==0.f){ m=m2; s=s2; return; }
  if (m2>m){ s = s*__expf(m-m2)+s2; m=m2; }
  else s += s2*__expf(m2-m);
}

__global__ __launch_bounds__(256) void k_attn_stats(const float* __restrict__ A_in,
    const float* __restrict__ asrc, const float* __restrict__ adst,
    float* __restrict__ mrow, float* __restrict__ srow)
{
  int i = blockIdx.x, tid = threadIdx.x;
  float ad0 = adst[2*i], ad1 = adst[2*i+1];
  float m0=-3.4e38f,s0=0.f,m1=-3.4e38f,s1=0.f;
  for (int j=tid; j<kN; j+=256){
    bool msk = (A_in[(size_t)i*kN+j] > 0.f) || (j==i);
    if (!msk) continue;
    float e0 = lrelu(asrc[2*j]+ad0);
    float e1 = lrelu(asrc[2*j+1]+ad1);
    onmerge(m0,s0,e0,1.f);
    onmerge(m1,s1,e1,1.f);
  }
  __shared__ float sm0[256], ss0[256], sm1[256], ss1[256];
  sm0[tid]=m0; ss0[tid]=s0; sm1[tid]=m1; ss1[tid]=s1; __syncthreads();
  for(int off=128; off>0; off>>=1){
    if(tid<off){
      float a=sm0[tid],b=ss0[tid]; onmerge(a,b,sm0[tid+off],ss0[tid+off]); sm0[tid]=a; ss0[tid]=b;
      float c=sm1[tid],d=ss1[tid]; onmerge(c,d,sm1[tid+off],ss1[tid+off]); sm1[tid]=c; ss1[tid]=d;
    }
    __syncthreads();
  }
  if(tid==0){ mrow[2*i]=sm0[0]; srow[2*i]=ss0[0]; mrow[2*i+1]=sm1[0]; srow[2*i+1]=ss1[0]; }
}

__global__ __launch_bounds__(256) void k_attn_p(const float* __restrict__ A_in,
    const float* __restrict__ asrc, const float* __restrict__ adst,
    const float* __restrict__ mrow, const float* __restrict__ srow,
    __bf16* __restrict__ P0, __bf16* __restrict__ P1)
{
  int j = blockIdx.x*256+threadIdx.x, i = blockIdx.y;
  size_t ix = (size_t)i*kN + j;
  bool msk = (A_in[ix] > 0.f) || (i==j);
  float p0=0.f, p1=0.f;
  if (msk){
    float e0 = lrelu(asrc[2*j]+adst[2*i]);
    float e1 = lrelu(asrc[2*j+1]+adst[2*i+1]);
    p0 = __expf(e0-mrow[2*i])/srow[2*i];
    p1 = __expf(e1-mrow[2*i+1])/srow[2*i+1];
  }
  P0[ix]=(__bf16)p0; P1[ix]=(__bf16)p1;
}

// ---------------- batchnorm / act ----------------
__global__ __launch_bounds__(256) void k_colstats(const float* __restrict__ X, int ld,
    float* __restrict__ mean, float* __restrict__ inv)
{
  int c = blockIdx.x, tid = threadIdx.x;
  __shared__ float sh[256];
  float s=0.f,q=0.f;
  for(int r=tid;r<kN;r+=256){ float v=X[(size_t)r*ld+c]; s+=v; q+=v*v; }
  s = blockReduceSum(s, sh);
  q = blockReduceSum(q, sh);
  if(tid==0){ float m=s/kN; float var=q/kN - m*m; mean[c]=m; inv[c]=rsqrtf(var+1e-5f); }
}

__global__ __launch_bounds__(256) void k_bn_act(const float* __restrict__ X, float* __restrict__ Y,
    int ld, const float* __restrict__ mean, const float* __restrict__ inv,
    const float* __restrict__ g, const float* __restrict__ b)
{
  size_t idx = (size_t)blockIdx.x*256+threadIdx.x;
  int c = (int)(idx % ld);
  float v = (X[idx]-mean[c])*inv[c]*g[c]+b[c];
  Y[idx] = eluf(v);
}

__global__ __launch_bounds__(256) void k_combine(const float* __restrict__ hA,
    const float* __restrict__ hM, const float* __restrict__ mu, float* __restrict__ h)
{
  size_t idx = (size_t)blockIdx.x*256+threadIdx.x;
  float sp = log1pf(__expf(mu[0]));
  h[idx] = hA[idx] + sp*hM[idx];
}

// ---------------- top-k / symmetrize ----------------
__global__ __launch_bounds__(256) void k_topk_row(const float* __restrict__ A,
    float* __restrict__ out, int n)
{
  int row = blockIdx.x, tid = threadIdx.x;
  __shared__ float bv[256]; __shared__ int bi[256];
  int chosen[8]; float chval[8];
  for(int it=0; it<8; it++){
    float best=-3.4e38f; int bidx=-1;
    for(int j=tid;j<n;j+=256){
      float v=A[(size_t)row*n+j];
      bool skip=false;
      for(int t=0;t<it;t++) if(chosen[t]==j) skip=true;
      if(skip) continue;
      if(v>best || (v==best && (bidx==-1 || j<bidx))){ best=v; bidx=j; }
    }
    bv[tid]=best; bi[tid]=bidx; __syncthreads();
    for(int off=128; off>0; off>>=1){
      if(tid<off){
        float v2=bv[tid+off]; int j2=bi[tid+off];
        if(j2!=-1 && (v2>bv[tid] || (v2==bv[tid] && (bi[tid]==-1 || j2<bi[tid])))){ bv[tid]=v2; bi[tid]=j2; }
      }
      __syncthreads();
    }
    chosen[it]=bi[0]; chval[it]=bv[0];
    __syncthreads();
  }
  for(int j=tid;j<n;j+=256) out[(size_t)row*n+j]=0.f;
  __syncthreads();
  if(tid<8 && chosen[tid]>=0) out[(size_t)row*n+chosen[tid]]=chval[tid];
}

__global__ __launch_bounds__(256) void k_symmax(const float* __restrict__ Sin,
    float* __restrict__ Mout, int n)
{
  size_t idx = (size_t)blockIdx.x*256+threadIdx.x;
  int i = (int)(idx / n), j = (int)(idx % n);
  Mout[idx] = (i==j) ? 0.f : fmaxf(Sin[(size_t)i*n+j], Sin[(size_t)j*n+i]);
}

// ---------------- row sums / scales ----------------
__global__ __launch_bounds__(256) void k_rowsum(const float* __restrict__ W,
    const float* __restrict__ cs, float* __restrict__ out)
{
  int i = blockIdx.x, tid = threadIdx.x;
  __shared__ float sh[256];
  float s=0.f;
  for(int j=tid;j<kN;j+=256){ float v=W[(size_t)i*kN+j]; if(cs) v*=cs[j]; s+=v; }
  s = blockReduceSum(s, sh);
  if(tid==0) out[i]=s;
}

__global__ __launch_bounds__(256) void k_dinv(const float* __restrict__ deg, float add, float* __restrict__ out){
  int i = blockIdx.x*256+threadIdx.x;
  float d = deg[i]+add;
  out[i] = d>0.f ? rsqrtf(d) : 0.f;
}

__global__ __launch_bounds__(256) void k_f32bf16(const float* __restrict__ src, __bf16* __restrict__ dst){
  size_t i = (size_t)blockIdx.x*256+threadIdx.x;
  dst[i] = (__bf16)src[i];
}

__global__ __launch_bounds__(256) void k_rowscale64(const float* __restrict__ in,
    const float* __restrict__ sc, float* __restrict__ out)
{
  size_t idx = (size_t)blockIdx.x*256+threadIdx.x;
  int i = (int)(idx >> 6);
  out[idx] = in[idx]*sc[i];
}

// ---------------- prune/rewire ----------------
__global__ __launch_bounds__(256) void k_pairvec(const float* __restrict__ h,
    const float* __restrict__ wp, const float* __restrict__ wr,
    float* up, float* vp, float* ur, float* vr)
{
  int i = blockIdx.x, tid = threadIdx.x;
  __shared__ float sh[256];
  float hv = h[(size_t)i*kHD+tid];
  float s;
  s = blockReduceSum(hv*wp[tid],     sh); if(tid==0) up[i]=s;
  s = blockReduceSum(hv*wp[256+tid], sh); if(tid==0) vp[i]=s;
  s = blockReduceSum(hv*wr[tid],     sh); if(tid==0) ur[i]=s;
  s = blockReduceSum(hv*wr[256+tid], sh); if(tid==0) vr[i]=s;
}

__global__ __launch_bounds__(256) void k_rewire_top2(const float* __restrict__ A_in,
    const float* __restrict__ M_hat, const float* __restrict__ coords,
    const float* __restrict__ ur, const float* __restrict__ vr,
    const float* __restrict__ wr, const float* __restrict__ br,
    const float* __restrict__ tau_p, int* __restrict__ keepidx)
{
  int i = blockIdx.x, tid = threadIdx.x;
  float tc = fmaxf(tau_p[0], 0.1f);
  float w4 = wr[512], bb = br[0];
  float cx = coords[2*i], cy = coords[2*i+1];
  float v0=-1e30f, v1=-1e30f; int i0=-1, i1=-1;
  for(int j=tid;j<kN;j+=256){
    if(j==i) continue;
    float aij = A_in[(size_t)i*kN+j];
    float d = fabsf(cx-coords[2*j])+fabsf(cy-coords[2*j+1]);
    if(!(d>0.f && d<=kRADIUS && aij<1e-6f)) continue;
    int lo = (i<j)?i:j, hi = (i<j)?j:i;
    float lr = ur[lo]+vr[hi]+w4*M_hat[(size_t)i*kN+j]+bb;
    float z = hcgate(lr, tc);
    if (z>v0 || (z==v0 && (i0==-1 || j<i0))){ v1=v0;i1=i0; v0=z;i0=j; }
    else if (z>v1 || (z==v1 && (i1==-1 || j<i1))){ v1=z;i1=j; }
  }
  __shared__ float s0[256], s1[256]; __shared__ int t0[256], t1[256];
  s0[tid]=v0; s1[tid]=v1; t0[tid]=i0; t1[tid]=i1; __syncthreads();
  for(int off=128; off>0; off>>=1){
    if(tid<off){
      float a0=s0[tid],a1=s1[tid]; int x0=t0[tid],x1=t1[tid];
      float b0v=s0[tid+off],b1v=s1[tid+off]; int y0=t0[tid+off],y1=t1[tid+off];
      if (y0!=-1 && (b0v>a0 || (b0v==a0 && (x0==-1 || y0<x0)))){ a1=a0;x1=x0; a0=b0v;x0=y0; }
      else if (y0!=-1 && (b0v>a1 || (b0v==a1 && (x1==-1 || y0<x1)))){ a1=b0v;x1=y0; }
      if (y1!=-1 && (b1v>a1 || (b1v==a1 && (x1==-1 || y1<x1)))){ a1=b1v;x1=y1; }
      s0[tid]=a0;s1[tid]=a1;t0[tid]=x0;t1[tid]=x1;
    }
    __syncthreads();
  }
  if(tid==0){
    keepidx[2*i]   = (s0[0]>-0.5f)? t0[0] : -1;
    keepidx[2*i+1] = (s1[0]>-0.5f)? t1[0] : -1;
  }
}

__global__ __launch_bounds__(256) void k_refine(const float* __restrict__ A_in,
    const float* __restrict__ M_hat, const float* __restrict__ coords,
    const float* __restrict__ up, const float* __restrict__ vp,
    const float* __restrict__ ur, const float* __restrict__ vr,
    const int* __restrict__ keepidx,
    const float* __restrict__ wp, const float* __restrict__ bp,
    const float* __restrict__ wr, const float* __restrict__ br,
    const float* __restrict__ tau_p,
    float* __restrict__ A_ref, float* __restrict__ acc)
{
  int j = blockIdx.x*256+threadIdx.x, i = blockIdx.y;
  float tc = fmaxf(tau_p[0], 0.1f);
  const float shift = logf(0.1f/1.1f);  // log(-gamma/zeta)
  size_t ix = (size_t)i*kN + j;
  float aij = A_in[ix];
  float m   = M_hat[ix];
  float out = 0.f;
  float c_p=0.f, c_ne=0.f, c_r=0.f, c_nc=0.f;
  if (i != j){
    int lo = (i<j)?i:j, hi = (i<j)?j:i;
    if (aij > 0.f){
      float lp = up[lo]+vp[hi]+wp[512]*m+bp[0];
      out = aij*hcgate(lp, tc);
      if (i<j){ c_p = sigm(lp-shift); c_ne=1.f; }
    } else {
      float d = fabsf(coords[2*i]-coords[2*j])+fabsf(coords[2*i+1]-coords[2*j+1]);
      if (d>0.f && d<=kRADIUS){
        float lr = ur[lo]+vr[hi]+wr[512]*m+br[0];
        float z = hcgate(lr, tc);
        bool keep = (keepidx[2*lo]==hi)||(keepidx[2*lo+1]==hi)||(keepidx[2*hi]==lo)||(keepidx[2*hi+1]==lo);
        if (keep) out = kETA*z;
        if (i<j){ c_r = sigm(lr-shift); c_nc=1.f; }
      }
    }
  }
  A_ref[ix] = out;
  __shared__ float sh[256];
  float t;
  t = blockReduceSum(c_p,  sh); if(threadIdx.x==0 && t!=0.f) atomicAdd(&acc[0],t);
  t = blockReduceSum(c_ne, sh); if(threadIdx.x==0 && t!=0.f) atomicAdd(&acc[1],t);
  t = blockReduceSum(c_r,  sh); if(threadIdx.x==0 && t!=0.f) atomicAdd(&acc[2],t);
  t = blockReduceSum(c_nc, sh); if(threadIdx.x==0 && t!=0.f) atomicAdd(&acc[3],t);
}

// ---------------- pooling / losses ----------------
__global__ __launch_bounds__(64) void k_softmax64(const float* __restrict__ PL,
    const float* __restrict__ bpool, float* __restrict__ S)
{
  int i = blockIdx.x, tid = threadIdx.x;
  __shared__ float sh[64];
  float v = PL[(size_t)i*kK+tid]+bpool[tid];
  sh[tid]=v; __syncthreads();
  for(int off=32; off>0; off>>=1){ if(tid<off) sh[tid]=fmaxf(sh[tid],sh[tid+off]); __syncthreads(); }
  float mx = sh[0]; __syncthreads();
  float e = __expf(v-mx);
  sh[tid]=e; __syncthreads();
  for(int off=32; off>0; off>>=1){ if(tid<off) sh[tid]+=sh[tid+off]; __syncthreads(); }
  S[(size_t)i*kK+tid] = e/sh[0];
}

// C[m,n] = sum_k A[k,m]*ks[k]*B[k,n]; one block per m
__global__ __launch_bounds__(256) void k_atb(const float* __restrict__ A, int lda,
    const float* __restrict__ B, int ldb, float* __restrict__ Cm, int ldc,
    int Ncols, int Kd, const float* __restrict__ kscale, int zero_diag)
{
  int m = blockIdx.x, n = threadIdx.x;
  __shared__ float Ash[256];
  float acc = 0.f;
  for(int k0=0; k0<Kd; k0+=256){
    float a = A[(size_t)(k0+threadIdx.x)*lda+m];
    if (kscale) a *= kscale[k0+threadIdx.x];
    Ash[threadIdx.x]=a; __syncthreads();
    if (n < Ncols){
      for(int t=0;t<256;t++) acc += Ash[t]*B[(size_t)(k0+t)*ldb+n];
    }
    __syncthreads();
  }
  if (n < Ncols){
    if (zero_diag && m==n) acc = 0.f;
    Cm[(size_t)m*ldc+n] = acc;
  }
}

__global__ __launch_bounds__(256) void k_prepcut(const float* __restrict__ gp,
    const float* __restrict__ dinv2, const float* __restrict__ dinvm,
    const float* __restrict__ ws1_, const float* __restrict__ ws2_,
    float* __restrict__ degcut, float* __restrict__ d2g, float* __restrict__ dmg)
{
  int i = blockIdx.x*256+threadIdx.x;
  float g = sigm(gp[0]);
  degcut[i] = (1.f-g)*dinv2[i]*ws1_[i] + g*dinvm[i]*ws2_[i];
  d2g[i] = (1.f-g)*dinv2[i];
  dmg[i] = g*dinvm[i];
}

__global__ __launch_bounds__(256) void k_dot(const float* __restrict__ X,
    const float* __restrict__ Y, int len, float* __restrict__ accp)
{
  int idx = blockIdx.x*256+threadIdx.x;
  float v = (idx<len) ? X[idx]*Y[idx] : 0.f;
  __shared__ float sh[256];
  v = blockReduceSum(v, sh);
  if(threadIdx.x==0) atomicAdd(accp, v);
}

__global__ __launch_bounds__(256) void k_colmean(const float* __restrict__ S, float* __restrict__ p){
  int c = blockIdx.x, tid = threadIdx.x;
  __shared__ float sh[256];
  float s=0.f; for(int r=tid;r<kN;r+=256) s += S[(size_t)r*kK+c];
  s = blockReduceSum(s, sh);
  if(tid==0) p[c] = s/kN;
}

__global__ __launch_bounds__(256) void k_motifred(const float* __restrict__ A_ref,
    const float* __restrict__ M_hat, const float* __restrict__ dinv2,
    const float* __restrict__ dinvm, float* __restrict__ acc)
{
  int j = blockIdx.x*256+threadIdx.x, i = blockIdx.y;
  size_t ix = (size_t)i*kN+j;
  float a = dinv2[i]*A_ref[ix]*dinv2[j];
  float m = dinvm[i]*M_hat[ix]*dinvm[j];
  __shared__ float sh[256];
  float t;
  t = blockReduceSum(a*m, sh); if(threadIdx.x==0 && t!=0.f) atomicAdd(&acc[5],t);
  t = blockReduceSum(a*a, sh); if(threadIdx.x==0 && t!=0.f) atomicAdd(&acc[6],t);
  t = blockReduceSum(m*m, sh); if(threadIdx.x==0 && t!=0.f) atomicAdd(&acc[7],t);
}

__global__ __launch_bounds__(256) void k_finalize(const float* __restrict__ StDS,
    const float* __restrict__ pvec, const float* __restrict__ degcut,
    const float* __restrict__ acc, float* __restrict__ out_total)
{
  __shared__ float sh[256];
  int tid = threadIdx.x;
  float s=0.f; for(int i2=tid;i2<kN;i2+=256) s+=degcut[i2];
  float sumdeg = blockReduceSum(s, sh);
  float trp=0.f; for(int m=tid;m<kK;m+=256) trp += StDS[m*kK+m];
  float tr = blockReduceSum(trp, sh);
  float o=0.f;
  for(int idx=tid; idx<kK*kK; idx+=256){
    int m=idx>>6, n2=idx&63;
    float t = StDS[idx]/(sumdeg+1e-8f) - ((m==n2)?(1.f/kK):0.f);
    o += t*t;
  }
  float L_ortho = blockReduceSum(o, sh);
  float bl=0.f;
  for(int c=tid;c<kK;c+=256){ float p=pvec[c]; bl -= p*logf(fmaxf(p,1e-8f)); }
  float L_bal = blockReduceSum(bl, sh);
  if(tid==0){
    float sSC1 = acc[4];
    float L_cut = (tr - sSC1)/(tr + 1e-8f);
    float ne = fmaxf(acc[1], 1.f), nc = acc[3];
    float L0p = acc[0]/ne;
    float L0r = (nc>0.f) ? acc[2]/fmaxf(nc,1.f) : 0.f;
    float L_motif = -acc[5]/(sqrtf(acc[6])*sqrtf(acc[7]) + 1e-8f);
    out_total[0] = 1.0f*L_cut + 0.5f*L_motif + 0.1f*(L0p+L0r) + 1.0f*L_ortho + 0.1f*L_bal;
  }
}

// ---------------- host launch ----------------
extern "C" void kernel_launch(void* const* d_in, const int* in_sizes, int n_in,
                              void* d_out, int out_size, void* d_ws, size_t ws_size,
                              hipStream_t stream) {
  (void)in_sizes; (void)n_in; (void)out_size; (void)ws_size;
  const float* x        = (const float*)d_in[0];
  const float* A_in     = (const float*)d_in[1];
  const float* A_motif  = (const float*)d_in[2];
  const float* coords   = (const float*)d_in[3];
  const float* W_gat    = (const float*)d_in[4];
  const float* att_src  = (const float*)d_in[5];
  const float* att_dst  = (const float*)d_in[6];
  const float* W_m      = (const float*)d_in[8];
  const float* bnA_g    = (const float*)d_in[10];
  const float* bnA_b    = (const float*)d_in[11];
  const float* bnM_g    = (const float*)d_in[12];
  const float* bnM_b    = (const float*)d_in[13];
  const float* mu       = (const float*)d_in[14];
  const float* tau      = (const float*)d_in[15];
  const float* w_prune  = (const float*)d_in[16];
  const float* b_prune  = (const float*)d_in[17];
  const float* w_rewire = (const float*)d_in[18];
  const float* b_rewire = (const float*)d_in[19];
  const float* W_pool   = (const float*)d_in[20];
  const float* b_pool   = (const float*)d_in[21];
  const float* gamma_pl = (const float*)d_in[22];

  const size_t NN = (size_t)kN*kN;
  float* w     = (float*)d_ws;
  float* buf0  = w;               // NN  (S_raw -> P0/P1 bf16 -> Abf/Mbf bf16)
  float* M_hat = w + NN;          // NN
  float* A_ref = w + 2*NN;        // NN
  float* xh    = w + 3*NN;        // N*256
  float* xm    = xh + (size_t)kN*kHD;
  float* hg    = xm + (size_t)kN*kHD;
  float* hA    = hg + (size_t)kN*kHD;
  float* hM    = hA + (size_t)kN*kHD;
  float* hh    = hM + (size_t)kN*kHD;
  float* T1    = hh + (size_t)kN*kHD;   // N*64 each below
  float* PL    = T1 + (size_t)kN*kK;
  float* Sm    = PL + (size_t)kN*kK;
  float* Ss1   = Sm + (size_t)kN*kK;
  float* Ssm   = Ss1 + (size_t)kN*kK;
  float* C1    = Ssm + (size_t)kN*kK;
  float* Rm    = C1 + (size_t)kN*kK;
  float* asrc  = Rm + (size_t)kN*kK;    // 2N each
  float* adst  = asrc + 2*kN;
  float* mrow  = adst + 2*kN;
  float* srow  = mrow + 2*kN;
  float* degm  = srow + 2*kN;           // N each
  float* dinvm = degm + kN;
  float* degr  = dinvm + kN;
  float* d1    = degr + kN;
  float* dinv2 = d1 + kN;
  float* ws1   = dinv2 + kN;
  float* ws2   = ws1 + kN;
  float* degcut= ws2 + kN;
  float* d2g   = degcut + kN;
  float* dmg   = d2g + kN;
  float* up    = dmg + kN;
  float* vp    = up + kN;
  float* ur    = vp + kN;
  float* vr    = ur + kN;
  float* meanA = vr + kN;   // 256 each
  float* invA  = meanA + 256;
  float* meanM = invA + 256;
  float* invM  = meanM + 256;
  float* pvec  = invM + 256;           // 64
  float* StDS  = pvec + 64;            // 4096
  float* Ac    = StDS + 4096;
  float* AcS   = Ac + 4096;
  float* accv  = AcS + 4096;           // 16
  int* keepidx = (int*)(accv + 16);    // 2N ints

  __bf16* P0  = (__bf16*)buf0;
  __bf16* P1  = P0 + NN;
  __bf16* Abf = (__bf16*)buf0;
  __bf16* Mbf = Abf + NN;
  __bf16* Mbf_early = (__bf16*)buf0;

  float* Xc      = (float*)d_out;
  float* Acoarse = Xc + kK*kHD;
  float* total   = Acoarse + kK*kK;

  hipMemsetAsync(accv, 0, 16*sizeof(float), stream);

  // input projections (WMMA, fp32->bf16 A)
  k_gemm<float><<<dim3(kHD/64, kN/64), 256, 0, stream>>>(x, kC, W_gat, kHD, xh, kHD, kC, nullptr, nullptr, 0, 0);
  k_gemm<float><<<dim3(kHD/64, kN/64), 256, 0, stream>>>(x, kC, W_m,   kHD, xm, kHD, kC, nullptr, nullptr, 0, 0);
  k_attvec<<<kN, 256, 0, stream>>>(xh, att_src, att_dst, asrc, adst);

  // motif branch: M_hat = row_topk(A_motif,8) sym; h_M = elu(BN(sym_norm(M_hat)@xm))
  k_topk_row<<<kN, 256, 0, stream>>>(A_motif, buf0, kN);
  k_symmax<<<(unsigned)(NN/256), 256, 0, stream>>>(buf0, M_hat, kN);
  k_rowsum<<<kN, 256, 0, stream>>>(M_hat, nullptr, degm);
  k_dinv<<<kN/256, 256, 0, stream>>>(degm, 0.f, dinvm);
  k_f32bf16<<<(unsigned)(NN/256), 256, 0, stream>>>(M_hat, Mbf_early);
  k_gemm<__bf16><<<dim3(kHD/64, kN/64), 256, 0, stream>>>(Mbf_early, kN, xm, kHD, hM, kHD, kN, dinvm, dinvm, 0, 0);
  k_colstats<<<kHD, 256, 0, stream>>>(hM, kHD, meanM, invM);
  k_bn_act<<<kN*kHD/256, 256, 0, stream>>>(hM, hM, kHD, meanM, invM, bnM_g, bnM_b);

  // GAT branch: softmax over masked neighbors, P bf16, then WMMA aggregation per head
  k_attn_stats<<<kN, 256, 0, stream>>>(A_in, asrc, adst, mrow, srow);
  k_attn_p<<<dim3(kN/256, kN), 256, 0, stream>>>(A_in, asrc, adst, mrow, srow, P0, P1);
  k_gemm<__bf16><<<dim3(128/64, kN/64), 256, 0, stream>>>(P0, kN, xh,      kHD, hg,      kHD, kN, nullptr, nullptr, 0, 0);
  k_gemm<__bf16><<<dim3(128/64, kN/64), 256, 0, stream>>>(P1, kN, xh+128,  kHD, hg+128,  kHD, kN, nullptr, nullptr, 0, 0);
  k_colstats<<<kHD, 256, 0, stream>>>(hg, kHD, meanA, invA);
  k_bn_act<<<kN*kHD/256, 256, 0, stream>>>(hg, hA, kHD, meanA, invA, bnA_g, bnA_b);
  k_combine<<<kN*kHD/256, 256, 0, stream>>>(hA, hM, mu, hh);

  // prune + rewire -> A_refined (fused with L0 reductions)
  k_pairvec<<<kN, 256, 0, stream>>>(hh, w_prune, w_rewire, up, vp, ur, vr);
  k_rewire_top2<<<kN, 256, 0, stream>>>(A_in, M_hat, coords, ur, vr, w_rewire, b_rewire, tau, keepidx);
  k_refine<<<dim3(kN/256, kN), 256, 0, stream>>>(A_in, M_hat, coords, up, vp, ur, vr, keepidx,
                                                 w_prune, b_prune, w_rewire, b_rewire, tau, A_ref, accv);
  k_rowsum<<<kN, 256, 0, stream>>>(A_ref, nullptr, degr);
  k_dinv<<<kN/256, 256, 0, stream>>>(degr, 1.f, d1);     // for A_refined + I
  k_dinv<<<kN/256, 256, 0, stream>>>(degr, 0.f, dinv2);  // A_hat

  // pooling: S = softmax(sym_norm(A_ref+I) @ (h@W_pool) + b_pool)
  k_gemm<float><<<dim3(kK/64, kN/64), 256, 0, stream>>>(hh, kHD, W_pool, kK, T1, kK, kHD, nullptr, nullptr, 0, 0);
  k_f32bf16<<<(unsigned)(NN/256), 256, 0, stream>>>(A_ref, Abf);
  k_f32bf16<<<(unsigned)(NN/256), 256, 0, stream>>>(M_hat, Mbf);
  k_gemm<__bf16><<<dim3(kK/64, kN/64), 256, 0, stream>>>(Abf, kN, T1, kK, PL, kK, kN, d1, d1, 1, 0);
  k_softmax64<<<kN, 64, 0, stream>>>(PL, b_pool, Sm);

  // coarse outputs
  k_atb<<<kK, 256, 0, stream>>>(Sm, kK, hh, kHD, Xc, kHD, kHD, kN, nullptr, 0);            // X_coarse
  k_gemm<__bf16><<<dim3(kK/64, kN/64), 256, 0, stream>>>(Abf, kN, Sm, kK, Rm, kK, kN, nullptr, nullptr, 0, 0);
  k_atb<<<kK, 256, 0, stream>>>(Sm, kK, Rm, kK, Ac, kK, kK, kN, nullptr, 1);               // A_c (diag zeroed)
  k_topk_row<<<kK, 256, 0, stream>>>(Ac, AcS, kK);
  k_symmax<<<(kK*kK)/256, 256, 0, stream>>>(AcS, Acoarse, kK);

  // losses
  k_rowsum<<<kN, 256, 0, stream>>>(A_ref, dinv2, ws1);
  k_rowsum<<<kN, 256, 0, stream>>>(M_hat, dinvm, ws2);
  k_prepcut<<<kN/256, 256, 0, stream>>>(gamma_pl, dinv2, dinvm, ws1, ws2, degcut, d2g, dmg);
  k_rowscale64<<<kN*kK/256, 256, 0, stream>>>(Sm, dinv2, Ss1);
  k_rowscale64<<<kN*kK/256, 256, 0, stream>>>(Sm, dinvm, Ssm);
  k_gemm<__bf16><<<dim3(kK/64, kN/64), 256, 0, stream>>>(Abf, kN, Ss1, kK, C1, kK, kN, nullptr, d2g, 0, 0);
  k_gemm<__bf16><<<dim3(kK/64, kN/64), 256, 0, stream>>>(Mbf, kN, Ssm, kK, C1, kK, kN, nullptr, dmg, 0, 1);
  k_atb<<<kK, 256, 0, stream>>>(Sm, kK, Sm, kK, StDS, kK, kK, kN, degcut, 0);              // S^T D S
  k_dot<<<(kN*kK)/256, 256, 0, stream>>>(Sm, C1, kN*kK, &accv[4]);
  k_colmean<<<kK, 256, 0, stream>>>(Sm, pvec);
  k_motifred<<<dim3(kN/256, kN), 256, 0, stream>>>(A_ref, M_hat, dinv2, dinvm, accv);
  k_finalize<<<1, 256, 0, stream>>>(StDS, pvec, degcut, accv, total);
}
